// LSTM_34437047780047
// MI455X (gfx1250) — matrix operand order repaired and checked by reading, not compiled, not run on hardware
//
#include <hip/hip_runtime.h>
#include <hip/hip_bf16.h>

// ---------------------------------------------------------------------------
// LSTM on MI455X (gfx1250): persistent grid + WMMA bf16 + LDS-staged operands.
//
// Per step: 4 GEMMs (1024x1024)@(1024x1024) = 8.6 GFLOP, 64 serialized steps.
// - bf16 weights (8 MB) are L2-resident (192 MB L2); gate tiles interleaved so
//   each wave's A slice is one contiguous 128 KB region -> single base address,
//   all loads use immediate offsets.
// - Each block stages its h-slab (1024 k x 64 cols = 128 KB bf16) into LDS
//   once per step via GLOBAL_LOAD_ASYNC_TO_LDS_B128 (ASYNCcnt path).
// - Cell state c lives in VGPRs across all 64 steps.
// - Native v_tanh_f32 / v_rcp_f32 for the gate nonlinearities.
// ---------------------------------------------------------------------------

#define SEQ_LEN 64
#define HIDDEN  1024
#define BATCH   1024
#define NCLS    10

#define BM 64
#define BN 64
#define MT (HIDDEN / BM)          // 16
#define BT (BATCH / BN)           // 16
#define NBLOCKS (MT * BT)         // 256 persistent blocks
#define KC (HIDDEN / 32)          // 32 k-chunks (WMMA K=32)
#define NBT (BN / 16)             // 4 local 16-col tiles per block
#define HB_ELEMS (KC * 64 * 512)        // one h buffer: 1M bf16 = 2 MB
#define SLAB_ELEMS (KC * NBT * 512)     // LDS h slab: 64K bf16 = 128 KB
#define LDS_BYTES (SLAB_ELEMS * 2 + 4 * BM * 8)   // 128 KB + 2 KB (w,b pairs)

typedef __attribute__((ext_vector_type(16))) __bf16 v16bf;
typedef __attribute__((ext_vector_type(8)))  float  v8f;

#if __has_builtin(__builtin_amdgcn_global_load_async_to_lds_b128)
#define ASYNC_STAGE 1
typedef int v4i_t __attribute__((ext_vector_type(4)));
typedef __attribute__((address_space(1))) v4i_t gv4i;   // global b128 payload
typedef __attribute__((address_space(3))) v4i_t lv4i;   // LDS b128 payload
#endif

__device__ __forceinline__ __bf16 f2bf(float f) {
  unsigned u = __builtin_bit_cast(unsigned, f);
  u = (u + 0x7FFFu + ((u >> 16) & 1u)) >> 16;       // round-to-nearest-even
  unsigned short s = (unsigned short)u;
  return __builtin_bit_cast(__bf16, s);
}

#if __has_builtin(__builtin_amdgcn_tanhf)
__device__ __forceinline__ float act_tanh(float v) { return __builtin_amdgcn_tanhf(v); }
#else
__device__ __forceinline__ float act_tanh(float v) {
  return 2.0f * __builtin_amdgcn_rcpf(1.0f + __expf(-2.0f * v)) - 1.0f;
}
#endif
__device__ __forceinline__ float act_sigm(float v) {
  return __builtin_amdgcn_rcpf(1.0f + __expf(-v));
}

// ---------------------------------------------------------------------------
// Prep: fp32 (H,H) weights -> bf16 in WMMA A-matrix per-lane layout.
// Tile = 16(M) x 32(K) = 512 elems, lane*16 contiguous.  A-layout (contiguous k order):
// lane = half*16+m; element e -> k = 16*half + e.
// Gate-interleaved tile id = (mtile*KC + kc)*4 + gate  (one contiguous 128 KB
// slice per mtile -> single base pointer + immediate offsets in the main loop).
// ---------------------------------------------------------------------------
__global__ void conv_weights(const float* __restrict__ Wg, const float* __restrict__ Wi,
                             const float* __restrict__ Wf, const float* __restrict__ Wo,
                             __bf16* __restrict__ A) {
  const int NT = 4 * (HIDDEN / 16) * KC;            // 8192 tiles
  int gtile = (int)((blockIdx.x * blockDim.x + threadIdx.x) >> 5);
  int lane  = (int)(threadIdx.x & 31);
  if (gtile >= NT) return;
  int gate  = gtile & 3;
  int kc    = (gtile >> 2) & (KC - 1);
  int mtile = gtile >> 7;                           // / (4*KC)
  const float* W = (gate == 0) ? Wg : (gate == 1) ? Wi : (gate == 2) ? Wf : Wo;
  int hl   = lane >> 4;
  int mrow = lane & 15;
  __bf16* dst = A + (size_t)gtile * 512 + lane * 16;
  const float* src = W + (size_t)(mtile * 16 + mrow) * HIDDEN + kc * 32;
#pragma unroll
  for (int e = 0; e < 16; ++e) {
    int v = e >> 1, p = e & 1;
    int k = 16 * hl + e;
    dst[e] = f2bf(src[k]);
  }
}

// ---------------------------------------------------------------------------
// Persistent LSTM kernel: 256 blocks x 256 threads (8 wave32).
// Block (mt,bt): rows [mt*64,+64) x cols [bt*64,+64).
// Wave w: msub=w&3 (16 rows), bsub=w>>2 (2 of the 4 local N-tiles).
// ---------------------------------------------------------------------------
__global__ __launch_bounds__(256, 1) void lstm_persistent(
    const float* __restrict__ x,
    const float* __restrict__ Wgx, const float* __restrict__ Wix,
    const float* __restrict__ Wfx, const float* __restrict__ Wox,
    const float* __restrict__ bG, const float* __restrict__ bI,
    const float* __restrict__ bF, const float* __restrict__ bO,
    const __bf16* __restrict__ A,
    __bf16* __restrict__ hB,          // 2 x HB_ELEMS (double buffered)
    float* __restrict__ hOut,         // final h, (H,B) m-major fp32
    int* __restrict__ bar) {
  extern __shared__ char smem[];
  __bf16* sB = (__bf16*)smem;                         // 128 KB h slab
  float2* wb = (float2*)(smem + SLAB_ELEMS * 2);      // 4*64 (w,bias) pairs

  const int blk = (int)blockIdx.x;
  const int mt  = blk / BT;
  const int bt  = blk % BT;
  const int tid = (int)threadIdx.x;

  // Stage per-row (Wx, bias) pairs for the block's 64 rows x 4 gates.
  {
    int g = tid >> 6, ml = tid & 63;
    int m = mt * BM + ml;
    const float* Ws = (g == 0) ? Wgx : (g == 1) ? Wix : (g == 2) ? Wfx : Wox;
    const float* Bs = (g == 0) ? bG : (g == 1) ? bI : (g == 2) ? bF : bO;
    wb[g * BM + ml] = make_float2(Ws[m], Bs[m]);
  }
  __syncthreads();

  const int waveu = __builtin_amdgcn_readfirstlane((int)(threadIdx.x >> 5));
  const int lane  = (int)(threadIdx.x & 31);
  const int msub  = waveu & 3;                        // scalar
  const int bsub  = waveu >> 2;                       // scalar
  const int hl    = lane >> 4;
  const int ln    = lane & 15;

  const int m0      = mt * BM + msub * 16;            // scalar
  const int btlL0   = bsub * 2;                       // local tile, scalar
  const int btile0  = bt * NBT + btlL0;               // global tile, scalar
  const int b0      = btile0 * 16 + ln;
  const int b1      = b0 + 16;
  const int mtileG  = mt * 4 + msub;                  // scalar
  const int laneOff = lane * 16;                      // element offset in tile

  // One contiguous 128 KB gate-interleaved A slice for this wave's 16 rows.
  const __bf16* AS = A + (size_t)mtileG * (KC * 4 * 512);

  v8f c0v = {};                                       // cell state in VGPRs
  v8f c1v = {};

#pragma unroll 1
  for (int t = 0; t < SEQ_LEN; ++t) {
    const __bf16* rbufS = hB + (size_t)(t & 1) * HB_ELEMS;
    __bf16*       wbuf  = hB + (size_t)((t + 1) & 1) * HB_ELEMS;

    // Stage h slab for this block's 4 local N-tiles (all k): 128 KB.
    // Per kc: 4 tiles x 1 KB contiguous in global; 256 threads x 16 B.
#ifdef ASYNC_STAGE
#pragma unroll 4
    for (int i = 0; i < KC; ++i) {
      __builtin_amdgcn_global_load_async_to_lds_b128(
          (gv4i*)(rbufS + (size_t)(i * 64 + bt * NBT) * 512 + tid * 8),
          (lv4i*)(sB + (size_t)i * (NBT * 512) + tid * 8), 0, 0);
    }
#if __has_builtin(__builtin_amdgcn_s_wait_asynccnt)
    __builtin_amdgcn_s_wait_asynccnt(0);
#else
    asm volatile("s_wait_asynccnt 0x0" ::: "memory");
#endif
#else
#pragma unroll 4
    for (int i = 0; i < KC; ++i) {
      *(float4*)(sB + (size_t)i * (NBT * 512) + tid * 8) =
          *(const float4*)(rbufS + (size_t)(i * 64 + bt * NBT) * 512 + tid * 8);
    }
#endif
    __syncthreads();

    // Accumulator init: rank-1 x-term + bias (INPUT_DIM == 1).
    float xb0 = x[b0 * SEQ_LEN + t];
    float xb1 = x[b1 * SEQ_LEN + t];
    v8f aG0, aG1, aI0, aI1, aF0, aF1, aO0, aO1;
#pragma unroll
    for (int j = 0; j < 8; ++j) {
      int ml = msub * 16 + j + 8 * hl;
      float2 pg = wb[0 * BM + ml];
      float2 pi = wb[1 * BM + ml];
      float2 pf = wb[2 * BM + ml];
      float2 po = wb[3 * BM + ml];
      aG0[j] = pg.x * xb0 + pg.y;  aG1[j] = pg.x * xb1 + pg.y;
      aI0[j] = pi.x * xb0 + pi.y;  aI1[j] = pi.x * xb1 + pi.y;
      aF0[j] = pf.x * xb0 + pf.y;  aF1[j] = pf.x * xb1 + pf.y;
      aO0[j] = po.x * xb0 + po.y;  aO1[j] = po.x * xb1 + po.y;
    }

#pragma unroll 4
    for (int kc = 0; kc < KC; ++kc) {
      v16bf bm0 = *(const v16bf*)(sB + (size_t)(kc * NBT + btlL0)     * 512 + laneOff);
      v16bf bm1 = *(const v16bf*)(sB + (size_t)(kc * NBT + btlL0 + 1) * 512 + laneOff);
      v16bf ag  = *(const v16bf*)(AS + (size_t)((kc * 4 + 0) * 512) + laneOff);
      v16bf ai  = *(const v16bf*)(AS + (size_t)((kc * 4 + 1) * 512) + laneOff);
      v16bf af  = *(const v16bf*)(AS + (size_t)((kc * 4 + 2) * 512) + laneOff);
      v16bf ao  = *(const v16bf*)(AS + (size_t)((kc * 4 + 3) * 512) + laneOff);
      aG0 = __builtin_amdgcn_wmma_f32_16x16x32_bf16(false, ag, false, bm0, (short)0, aG0, false, false);
      aG1 = __builtin_amdgcn_wmma_f32_16x16x32_bf16(false, ag, false, bm1, (short)0, aG1, false, false);
      aI0 = __builtin_amdgcn_wmma_f32_16x16x32_bf16(false, ai, false, bm0, (short)0, aI0, false, false);
      aI1 = __builtin_amdgcn_wmma_f32_16x16x32_bf16(false, ai, false, bm1, (short)0, aI1, false, false);
      aF0 = __builtin_amdgcn_wmma_f32_16x16x32_bf16(false, af, false, bm0, (short)0, aF0, false, false);
      aF1 = __builtin_amdgcn_wmma_f32_16x16x32_bf16(false, af, false, bm1, (short)0, aF1, false, false);
      aO0 = __builtin_amdgcn_wmma_f32_16x16x32_bf16(false, ao, false, bm0, (short)0, aO0, false, false);
      aO1 = __builtin_amdgcn_wmma_f32_16x16x32_bf16(false, ao, false, bm1, (short)0, aO1, false, false);
    }

    // Gate nonlinearities + state update + h store (B-layout bf16 tiles:
    // tile (k>>5, b>>4), lane' = (klocal>>4)*16 + n, element = klocal&15).
#pragma unroll
    for (int j = 0; j < 8; ++j) {
      int m = m0 + j + 8 * hl;
      int kct = m >> 5, kl = m & 31;
      size_t base = (size_t)(kct * 64) * 512 + (size_t)((kl >> 4) * 16 + ln) * 16 + (kl & 15);
      {
        float g  = act_tanh(aG0[j]);
        float ii = act_sigm(aI0[j]);
        float ff = act_sigm(aF0[j]);
        float oo = act_sigm(aO0[j]);
        float c  = g * ii + c0v[j] * ff;
        c0v[j] = c;
        float h = act_tanh(c) * oo;
        wbuf[base + (size_t)btile0 * 512] = f2bf(h);
        if (t == SEQ_LEN - 1) hOut[(size_t)m * BATCH + b0] = h;
      }
      {
        float g  = act_tanh(aG1[j]);
        float ii = act_sigm(aI1[j]);
        float ff = act_sigm(aF1[j]);
        float oo = act_sigm(aO1[j]);
        float c  = g * ii + c1v[j] * ff;
        c1v[j] = c;
        float h = act_tanh(c) * oo;
        wbuf[base + (size_t)(btile0 + 1) * 512] = f2bf(h);
        if (t == SEQ_LEN - 1) hOut[(size_t)m * BATCH + b1] = h;
      }
    }

    // Device-wide barrier between timesteps (monotonic counter, acq/rel).
    if (t != SEQ_LEN - 1) {
      __threadfence();
      __syncthreads();
      if (threadIdx.x == 0) {
        int target = (t + 1) * NBLOCKS;
        __hip_atomic_fetch_add(bar, 1, __ATOMIC_ACQ_REL, __HIP_MEMORY_SCOPE_AGENT);
        while (__hip_atomic_load(bar, __ATOMIC_ACQUIRE, __HIP_MEMORY_SCOPE_AGENT) < target) {
          __builtin_amdgcn_s_sleep(2);
        }
      }
      __syncthreads();
    }
  }
}

// ---------------------------------------------------------------------------
// Projection: out[b][c] = sum_m Wph[c][m] * h[m][b] + bp[c].  (20 MFLOP.)
// ---------------------------------------------------------------------------
__global__ void proj_kernel(const float* __restrict__ h, const float* __restrict__ Wph,
                            const float* __restrict__ bp, float* __restrict__ out) {
  __shared__ float red[NCLS][256];
  int b = (int)blockIdx.x;
  int tid = (int)threadIdx.x;
  float acc[NCLS];
#pragma unroll
  for (int c = 0; c < NCLS; ++c) acc[c] = 0.0f;
  for (int m = tid; m < HIDDEN; m += 256) {
    float hv = h[(size_t)m * BATCH + b];
#pragma unroll
    for (int c = 0; c < NCLS; ++c) acc[c] += Wph[c * HIDDEN + m] * hv;
  }
#pragma unroll
  for (int c = 0; c < NCLS; ++c) red[c][tid] = acc[c];
  __syncthreads();
  for (int s = 128; s > 0; s >>= 1) {
    if (tid < s) {
#pragma unroll
      for (int c = 0; c < NCLS; ++c) red[c][tid] += red[c][tid + s];
    }
    __syncthreads();
  }
  if (tid < NCLS) out[(size_t)b * NCLS + tid] = red[tid][0] + bp[tid];
}

// ---------------------------------------------------------------------------
extern "C" void kernel_launch(void* const* d_in, const int* in_sizes, int n_in,
                              void* d_out, int out_size, void* d_ws, size_t ws_size,
                              hipStream_t stream) {
  (void)in_sizes; (void)n_in; (void)out_size; (void)ws_size;
  const float* x   = (const float*)d_in[0];
  const float* Wgx = (const float*)d_in[1];
  const float* Wgh = (const float*)d_in[2];
  const float* Wix = (const float*)d_in[3];
  const float* Wih = (const float*)d_in[4];
  const float* Wfx = (const float*)d_in[5];
  const float* Wfh = (const float*)d_in[6];
  const float* Wox = (const float*)d_in[7];
  const float* Woh = (const float*)d_in[8];
  const float* Wph = (const float*)d_in[9];
  const float* bG  = (const float*)d_in[10];
  const float* bI  = (const float*)d_in[11];
  const float* bF  = (const float*)d_in[12];
  const float* bO  = (const float*)d_in[13];
  const float* bp  = (const float*)d_in[14];

  char* ws = (char*)d_ws;
  __bf16* Abuf = (__bf16*)(ws);                         // 8 MB  bf16 weights
  __bf16* hB   = (__bf16*)(ws + (size_t)(8u << 20));    // 4 MB  h double buffer
  float*  hOut = (float*)(ws + (size_t)(12u << 20));    // 4 MB  final h fp32
  int*    bar  = (int*)(ws + (size_t)(16u << 20));      // 4 B   grid barrier

  (void)hipMemsetAsync(hB, 0, (size_t)HB_ELEMS * sizeof(__bf16), stream);
  (void)hipMemsetAsync(bar, 0, sizeof(int), stream);

  conv_weights<<<(4 * (HIDDEN / 16) * KC * 32) / 256, 256, 0, stream>>>(
      Wgh, Wih, Wfh, Woh, Abuf);

  lstm_persistent<<<NBLOCKS, 256, LDS_BYTES, stream>>>(
      x, Wgx, Wix, Wfx, Wox, bG, bI, bF, bO, Abuf, hB, hOut, bar);

  proj_kernel<<<BATCH, 256, 0, stream>>>(hOut, Wph, bp, (float*)d_out);
}
